// ChunkedConvblock_77043123356107
// MI455X (gfx1250) — matrix operand order repaired and checked by reading, not compiled, not run on hardware
//
#include <hip/hip_runtime.h>
#include <hip/hip_bf16.h>

typedef __bf16 bf16;
typedef __attribute__((ext_vector_type(16))) __bf16 v16bf;
typedef __attribute__((ext_vector_type(8)))  __bf16 v8bf;
typedef __attribute__((ext_vector_type(4)))  __bf16 v4bf;
typedef __attribute__((ext_vector_type(8)))  float  v8f;

#define HID     512
#define LCH     64          // rows per chunk
#define XPITCH  520         // bf16 elems (1040B, 16B aligned)
#define CTPITCH 520
#define CTTP    72          // CT-transposed pitch (144B, 16B aligned)
#define SPITCH  68
#define PPITCH  72

__device__ __forceinline__ v16bf frag_cat(v8bf lo, v8bf hi) {
  return __builtin_shufflevector(lo, hi, 0,1,2,3,4,5,6,7,8,9,10,11,12,13,14,15);
}

// A-matrix fragment (16x32 bf16): lane l -> row (rowBase + (l&15));
// K values = {g*8 + 0..7, g*8 + 16..23}, g = l>>4.  Two b128 loads.
__device__ __forceinline__ v16bf load_fragA(const bf16* buf, int pitch, int rowBase, int kb) {
  const int l = threadIdx.x & 31;
  const bf16* p = buf + (rowBase + (l & 15)) * pitch + kb + ((l >> 4) << 3);
  v8bf lo = *(const v8bf*)(p);
  v8bf hi = *(const v8bf*)(p + 16);
  return frag_cat(lo, hi);
}

// B-matrix fragment (32x16 bf16): lane l -> column (rowBase + (l&15)) stored
// as a row in the (N-major, K-contiguous) source; K values = {g*8 + 0..7, g*8 + 16..23}.
__device__ __forceinline__ v16bf load_fragB(const bf16* buf, int pitch, int rowBase, int kb) {
  const int l = threadIdx.x & 31;
  const bf16* p = buf + (rowBase + (l & 15)) * pitch + kb + ((l >> 4) << 3);
  v8bf lo = *(const v8bf*)(p);
  v8bf hi = *(const v8bf*)(p + 16);
  return frag_cat(lo, hi);
}

__device__ __forceinline__ v8f zero8() {
  v8f z = {0.f,0.f,0.f,0.f,0.f,0.f,0.f,0.f};
  return z;
}

#define WMMA_BF16(A,B,C) \
  __builtin_amdgcn_wmma_f32_16x16x32_bf16(false,(A),false,(B),(short)0,(C),false,false)

// ---------------------------------------------------------------------------
// Weight repack: conv_w [O=512][I=512][tap=2] f32  ->  wbf [tap][o][i] bf16
// (N-major, K-contiguous layout so WMMA B-fragments are two b128 loads)
// ---------------------------------------------------------------------------
__global__ __launch_bounds__(256) void convert_w_kernel(const float* __restrict__ w,
                                                        bf16* __restrict__ wbf) {
  const int idx = blockIdx.x * 256 + threadIdx.x;       // o*512 + i, 262144 total
  const float2 t = *(const float2*)(w + (size_t)idx * 2);
  wbf[idx]                 = (bf16)t.x;                 // tap 0
  wbf[HID * HID + idx]     = (bf16)t.y;                 // tap 1
}

// ---------------------------------------------------------------------------
// Fused chunk kernel: conv1d(k=2,dil=2)+relu -> attention -> LN -> mean-pool
// One workgroup (256 threads = 8 waves) per chunk of [64 x 512].
// ---------------------------------------------------------------------------
__global__ __launch_bounds__(256) void fused_chunk_kernel(
    const float* __restrict__ x, const bf16* __restrict__ wbf,
    const float* __restrict__ cb, const float* __restrict__ lg,
    const float* __restrict__ lb, float* __restrict__ out) {

  __shared__ bf16  Xp [66 * XPITCH];    // rows -1..64 of X (zero padded), bf16
  __shared__ bf16  CT [64 * CTPITCH];   // conv output, row-major [t][d]
  __shared__ bf16  CTt[HID * CTTP];     // conv output, transposed [d][t]
  __shared__ float Sc [64 * SPITCH];    // attention scores f32
  __shared__ bf16  Pbf[64 * PPITCH];    // softmax(P) bf16
  __shared__ float rstat[128];          // per-row (sum, sumsq)

  const int tid  = threadIdx.x;
  const int l    = tid & 31;
  const int wave = tid >> 5;
  const int g    = l >> 4;
  const int nl   = l & 15;
  const int ob   = wave * 64;           // this wave's output-column base
  const size_t chunk = blockIdx.x;

  // ---- Phase 0: stage X -> LDS (bf16), zero pad rows, zero LN stats ----
  const float4* xg = (const float4*)(x + chunk * (LCH * HID));
  #pragma unroll
  for (int it = 0; it < 32; ++it) {
    const int idx = tid + it * 256;               // 0..8191 float4s
    const float4 f = xg[idx];
    const int t = idx >> 7;                       // row (128 float4 per row)
    const int d = (idx << 2) & 511;
    v4bf v; v[0] = (bf16)f.x; v[1] = (bf16)f.y; v[2] = (bf16)f.z; v[3] = (bf16)f.w;
    *(v4bf*)&Xp[(t + 1) * XPITCH + d] = v;
  }
  for (int d = tid; d < HID; d += 256) {
    Xp[d] = (bf16)0.f;                            // X[-1] = 0
    Xp[65 * XPITCH + d] = (bf16)0.f;              // X[64] = 0
  }
  if (tid < 128) rstat[tid] = 0.f;
  __syncthreads();

  // ---- Phase 1: conv GEMM  C = relu(X[t-1] W0^T + X[t+1] W1^T + b) ----
  // wave handles all 4 M-tiles x its 4 N-tiles; K = 512 per tap.
  v8f acc[4][4];
  #pragma unroll
  for (int m = 0; m < 4; ++m)
    #pragma unroll
    for (int n = 0; n < 4; ++n) acc[m][n] = zero8();

  const bf16* w0 = wbf;
  const bf16* w1 = wbf + HID * HID;
  for (int kt = 0; kt < 16; ++kt) {
    const int kb = kt * 32;
    v16bf a0[4], a1[4];
    #pragma unroll
    for (int m = 0; m < 4; ++m) {
      a0[m] = load_fragA(Xp, XPITCH, m * 16 + 0, kb);   // Xp[t]   = X[t-1]
      a1[m] = load_fragA(Xp, XPITCH, m * 16 + 2, kb);   // Xp[t+2] = X[t+1]
    }
    #pragma unroll
    for (int n = 0; n < 4; ++n) {
      const v16bf b0 = load_fragB(w0, HID, ob + n * 16, kb);
      const v16bf b1 = load_fragB(w1, HID, ob + n * 16, kb);
      #pragma unroll
      for (int m = 0; m < 4; ++m) {
        acc[m][n] = WMMA_BF16(a0[m], b0, acc[m][n]);
        acc[m][n] = WMMA_BF16(a1[m], b1, acc[m][n]);
      }
    }
  }

  // conv epilogue: + bias, relu, store both CT (row-major) and CTt (transposed)
  #pragma unroll
  for (int n = 0; n < 4; ++n) {
    const int col = ob + n * 16 + nl;
    const float bias = cb[col];
    #pragma unroll
    for (int m = 0; m < 4; ++m) {
      #pragma unroll
      for (int v = 0; v < 8; ++v) {
        const int row = m * 16 + v + (g << 3);       // D layout: M = v + 8g
        float h = acc[m][n][v] + bias;
        h = fmaxf(h, 0.f);
        const bf16 hb = (bf16)h;
        CT [row * CTPITCH + col] = hb;
        CTt[col * CTTP    + row] = hb;
      }
    }
  }
  __syncthreads();

  // ---- Phase 2: scores = X @ CT^T / sqrt(D); 2 (m,n) tile-pairs per wave ----
  const int pr0 = wave * 2, pr1 = wave * 2 + 1;
  const int m0 = pr0 >> 2, n0 = pr0 & 3;
  const int m1 = pr1 >> 2, n1 = pr1 & 3;
  v8f s0 = zero8(), s1 = zero8();
  for (int kt = 0; kt < 16; ++kt) {
    const int kb = kt * 32;
    const v16bf aA = load_fragA(Xp, XPITCH, m0 * 16 + 1, kb);   // queries = X
    const v16bf bA = load_fragB(CT, CTPITCH, n0 * 16, kb);
    s0 = WMMA_BF16(aA, bA, s0);
    const v16bf aB = load_fragA(Xp, XPITCH, m1 * 16 + 1, kb);
    const v16bf bB = load_fragB(CT, CTPITCH, n1 * 16, kb);
    s1 = WMMA_BF16(aB, bB, s1);
  }
  const float sc = 0.04419417382415922f;   // 1/sqrt(512)
  #pragma unroll
  for (int v = 0; v < 8; ++v) {
    Sc[(m0 * 16 + v + (g << 3)) * SPITCH + n0 * 16 + nl] = s0[v] * sc;
    Sc[(m1 * 16 + v + (g << 3)) * SPITCH + n1 * 16 + nl] = s1[v] * sc;
  }
  __syncthreads();

  // ---- Phase 3: row softmax (64 rows, one thread per row) ----
  if (tid < 64) {
    float mx = -1e30f;
    for (int k = 0; k < 64; ++k) mx = fmaxf(mx, Sc[tid * SPITCH + k]);
    float sum = 0.f;
    for (int k = 0; k < 64; ++k) {
      const float e = __expf(Sc[tid * SPITCH + k] - mx);
      Sc[tid * SPITCH + k] = e;
      sum += e;
    }
    const float inv = 1.f / sum;
    for (int k = 0; k < 64; ++k)
      Pbf[tid * PPITCH + k] = (bf16)(Sc[tid * SPITCH + k] * inv);
  }
  __syncthreads();

  // ---- Phase 4: attn = P @ CT  (K = 64), B frags from transposed CTt ----
  v8f aat[4][4];
  #pragma unroll
  for (int m = 0; m < 4; ++m)
    #pragma unroll
    for (int n = 0; n < 4; ++n) aat[m][n] = zero8();

  #pragma unroll
  for (int kt = 0; kt < 2; ++kt) {
    const int kb = kt * 32;
    v16bf ap[4];
    #pragma unroll
    for (int m = 0; m < 4; ++m) ap[m] = load_fragA(Pbf, PPITCH, m * 16, kb);
    #pragma unroll
    for (int n = 0; n < 4; ++n) {
      const v16bf bb = load_fragB(CTt, CTTP, ob + n * 16, kb);
      #pragma unroll
      for (int m = 0; m < 4; ++m) aat[m][n] = WMMA_BF16(ap[m], bb, aat[m][n]);
    }
  }

  // ---- Phase 5a: LN statistics (h = attn + CT), cross-lane + LDS atomics ----
  #pragma unroll
  for (int m = 0; m < 4; ++m) {
    #pragma unroll
    for (int v = 0; v < 8; ++v) {
      const int row = m * 16 + v + (g << 3);
      float s = 0.f, q = 0.f;
      #pragma unroll
      for (int n = 0; n < 4; ++n) {
        const float h = aat[m][n][v] + (float)CT[row * CTPITCH + ob + n * 16 + nl];
        s += h; q += h * h;
      }
      #pragma unroll
      for (int off = 1; off < 16; off <<= 1) {
        s += __shfl_xor(s, off, 32);
        q += __shfl_xor(q, off, 32);
      }
      if (nl == 0) {
        atomicAdd(&rstat[row * 2 + 0], s);
        atomicAdd(&rstat[row * 2 + 1], q);
      }
    }
  }
  __syncthreads();

  // ---- Phase 5b: normalize + mean-pool over rows, write output ----
  float pooled[4] = {0.f, 0.f, 0.f, 0.f};
  #pragma unroll
  for (int m = 0; m < 4; ++m) {
    #pragma unroll
    for (int v = 0; v < 8; ++v) {
      const int row = m * 16 + v + (g << 3);
      const float mean = rstat[row * 2 + 0] * (1.f / 512.f);
      const float var  = rstat[row * 2 + 1] * (1.f / 512.f) - mean * mean;
      const float rs   = rsqrtf(var + 1e-5f);
      #pragma unroll
      for (int n = 0; n < 4; ++n) {
        const float h = aat[m][n][v] + (float)CT[row * CTPITCH + ob + n * 16 + nl];
        pooled[n] += (h - mean) * rs;
      }
    }
  }
  #pragma unroll
  for (int n = 0; n < 4; ++n) pooled[n] += __shfl_xor(pooled[n], 16, 32);
  if (l < 16) {
    #pragma unroll
    for (int n = 0; n < 4; ++n) {
      const int col = ob + n * 16 + l;
      out[chunk * HID + col] = lg[col] * (pooled[n] * (1.f / 64.f)) + lb[col];
    }
  }
}

// ---------------------------------------------------------------------------
extern "C" void kernel_launch(void* const* d_in, const int* in_sizes, int n_in,
                              void* d_out, int out_size, void* d_ws, size_t ws_size,
                              hipStream_t stream) {
  const float* x  = (const float*)d_in[0];   // [32,4096,512]
  const float* cw = (const float*)d_in[1];   // [512,512,2]
  const float* cb = (const float*)d_in[2];   // [512]
  const float* lg = (const float*)d_in[3];   // [512]
  const float* lb = (const float*)d_in[4];   // [512]
  (void)n_in; (void)out_size; (void)ws_size;

  bf16* wbf = (bf16*)d_ws;                   // [2][512][512] bf16 = 1 MB

  const int chunks = in_sizes[0] / (LCH * HID);   // 2048

  convert_w_kernel<<<(HID * HID) / 256, 256, 0, stream>>>(cw, wbf);
  fused_chunk_kernel<<<chunks, 256, 0, stream>>>(x, wbf, cb, lg, lb, (float*)d_out);
}
